// MeshEncoderDecoder_57415122812961
// MI455X (gfx1250) — matrix operand order repaired and checked by reading, not compiled, not run on hardware
//
#include <hip/hip_runtime.h>

typedef __attribute__((ext_vector_type(16))) __bf16 v16bf;
typedef __attribute__((ext_vector_type(8)))  __bf16 v8bf;
typedef __attribute__((ext_vector_type(8)))  float  v8f;

#define NE 50000
#define NB 2

// ---------------------------------------------------------------------------
// B-fragment assembly (wave32 WMMA 16x16x32 bf16, ISA 7.12.2):
//  B (32x16, KxN): lane L (0..15) col N=L holds K = k0+0..7, k0+16..23;
//                  lane L+16 col N=L holds K = k0+8..15, k0+24..31.
//  -> two contiguous 16B ds_load_b128 per lane.
// A-fragments come pre-swizzled in memory (repack_wmma_a_kernel): one
// contiguous 32B load per lane; a wave reads a contiguous 1KB block.
// ---------------------------------------------------------------------------
__device__ __forceinline__ v16bf frag16(const __bf16* p0, const __bf16* p1) {
  v8bf a = *(const v8bf*)p0;
  v8bf b = *(const v8bf*)p1;
  v16bf r;
#pragma unroll
  for (int i = 0; i < 8; ++i) { r[i] = a[i]; r[i + 8] = b[i]; }
  return r;
}

// Repack f32 weights (O x K row-major) into bf16 WMMA-A fragment-linear form:
// dst[((mt*KT + kt)*32 + lane)*16 + j] = w[m*K + k] with
//   m = mt*16 + (lane&15), half = lane>>4,
//   k = kt*32 + (j<8 ? half*8 + j : 16 + half*8 + (j-8)).
__global__ void repack_wmma_a_kernel(const float* __restrict__ w, __bf16* __restrict__ d,
                                     int O, int K) {
  long long idx = (long long)blockIdx.x * blockDim.x + threadIdx.x;
  if (idx >= (long long)O * K) return;
  int j = (int)(idx & 15);
  long long t = idx >> 4;
  int lane = (int)(t & 31); t >>= 5;
  int KT = K >> 5;
  int kt = (int)(t % KT), mt = (int)(t / KT);
  int m = mt * 16 + (lane & 15);
  int half = lane >> 4;
  int k = kt * 32 + ((j < 8) ? (half * 8 + j) : (16 + half * 8 + (j - 8)));
  d[idx] = (__bf16)w[(long long)m * K + k];
}

// Layer-1 mesh conv, tiny K (25 or 15): scalar VALU (0.6 GFLOP total).
__global__ void mesh1_kernel(const float* __restrict__ fe, const int* __restrict__ gm,
                             const float* __restrict__ w, const float* __restrict__ bias,
                             float* __restrict__ y, int C, int c_off, int E) {
  int e = blockIdx.x * blockDim.x + threadIdx.x;
  int b = blockIdx.z;
  if (e >= E) return;
  const float* xb = fe + ((long long)b * 8 + c_off) * E;
  const int* gp = gm + ((long long)b * E + e) * 4;
  int g0 = gp[0], g1 = gp[1], g2 = gp[2], g3 = gp[3];
  float G[5][5];
  for (int c = 0; c < C; ++c) {
    const float* xc = xb + (long long)c * E;
    float x0 = xc[e], f1 = xc[g0], f2 = xc[g1], f3 = xc[g2], f4 = xc[g3];
    G[c][0] = x0;
    G[c][1] = f1 + f3;
    G[c][2] = f2 + f4;
    G[c][3] = fabsf(f1 - f3);
    G[c][4] = fabsf(f2 - f4);
  }
  float* yb = y + (long long)b * 128 * E + e;
  for (int o = 0; o < 128; ++o) {
    float acc = bias[o];
    const float* wo = w + o * C * 5;
    for (int c = 0; c < C; ++c)
#pragma unroll
      for (int s = 0; s < 5; ++s)
        acc += G[c][s] * wo[c * 5 + s];
    yb[(long long)o * E] = acc;
  }
}

// Per-(b,channel) mean / rsqrt(var+eps) over E. One block per row.
__global__ void rowstat_kernel(const float* __restrict__ y, float* __restrict__ stats,
                               int E, float eps) {
  int row = blockIdx.x;
  const float* p = y + (long long)row * E;
  float s = 0.f, s2 = 0.f;
  for (int i = threadIdx.x; i < E; i += blockDim.x) {
    float v = p[i];
    s += v; s2 += v * v;
  }
  __shared__ float sh0[256], sh1[256];
  sh0[threadIdx.x] = s; sh1[threadIdx.x] = s2;
  __syncthreads();
  for (int o = 128; o > 0; o >>= 1) {
    if (threadIdx.x < o) { sh0[threadIdx.x] += sh0[threadIdx.x + o]; sh1[threadIdx.x] += sh1[threadIdx.x + o]; }
    __syncthreads();
  }
  if (threadIdx.x == 0) {
    float m = sh0[0] / (float)E;
    float var = sh1[0] / (float)E - m * m;
    stats[row * 2] = m;
    stats[row * 2 + 1] = rsqrtf(var + eps);
  }
}

// out = act((y - m) * rs [+ res]); mode 0: relu, 1: relu(+res), 2: sigmoid.
__global__ void inorm_act_kernel(const float* __restrict__ y, const float* __restrict__ stats,
                                 const float* __restrict__ res, float* __restrict__ out,
                                 long long out_bs, int E, int mode) {
  int row = blockIdx.y;
  int b = row >> 7, o = row & 127;
  float m = stats[row * 2], rs = stats[row * 2 + 1];
  long long ibase = (long long)row * E;
  long long obase = (long long)b * out_bs + (long long)o * E;
  for (int i = blockIdx.x * blockDim.x + threadIdx.x; i < E; i += gridDim.x * blockDim.x) {
    float v = (y[ibase + i] - m) * rs;
    if (mode == 1) v += res[ibase + i];
    if (mode == 2) v = 1.f / (1.f + __expf(-v));
    else           v = fmaxf(v, 0.f);
    out[obase + i] = v;
  }
}

// ---------------------------------------------------------------------------
// Pointwise channel GEMM with WMMA: Y[b,o,e] = sum_c W[o,c]*X[b,c,e] + bias.
// Block = 256 thr (8 waves), 128 elements/block.  Wave owns one M-tile and
// carries NACC accumulators so each A-fragment is reused NACC times.
// ---------------------------------------------------------------------------
template <int C, int O>
__global__ void conv1_wmma_kernel(const float* __restrict__ X, long long xbs,
                                  const __bf16* __restrict__ Wp, const float* __restrict__ bias,
                                  float* __restrict__ Y, long long ybs, int E) {
  constexpr int KP = C + 8;       // LDS pitch (bf16) -> conflict-free ds_load_b128
  constexpr int KT = C / 32;      // k-tiles
  constexpr int MT = O / 16;      // m-tiles
  constexpr int NG = 8 / MT;      // wave-groups across N
  constexpr int NACC = 8 / NG;    // n-tiles per wave
  __shared__ __bf16 Xs[128 * KP];

  int b = blockIdx.z;
  int e0 = blockIdx.x * 128;
  const float* Xb = X + (long long)b * xbs;
  for (int idx = threadIdx.x; idx < 128 * C; idx += blockDim.x) {
    int c = idx >> 7, n = idx & 127;
    int e = e0 + n;
    Xs[n * KP + c] = (e < E) ? (__bf16)Xb[(long long)c * E + e] : (__bf16)0.f;
  }
  __syncthreads();

  int wave = threadIdx.x >> 5;
  int lane = threadIdx.x & 31;
  int ln = lane & 15, half = lane >> 4;
  int mt = wave % MT, ng = wave / MT;

  v8f acc[NACC];
#pragma unroll
  for (int j = 0; j < NACC; ++j)
#pragma unroll
    for (int i = 0; i < 8; ++i) acc[j][i] = 0.f;

#pragma unroll
  for (int kk = 0; kk < KT; ++kk) {
    v16bf A = *(const v16bf*)(Wp + (((long long)mt * KT + kk) * 32 + lane) * 16);
#pragma unroll
    for (int j = 0; j < NACC; ++j) {
      int nt = ng * NACC + j;
      const __bf16* xr = &Xs[(nt * 16 + ln) * KP + kk * 32 + half * 8];
      v16bf Bm = frag16(xr, xr + 16);
      acc[j] = __builtin_amdgcn_wmma_f32_16x16x32_bf16(false, A, false, Bm, (short)0, acc[j], false, false);
    }
  }

  float* Yb = Y + (long long)b * ybs;
#pragma unroll
  for (int j = 0; j < NACC; ++j) {
    int nt = ng * NACC + j;
    int e = e0 + nt * 16 + ln;
    if (e < E) {
#pragma unroll
      for (int i = 0; i < 8; ++i) {
        int m = mt * 16 + half * 8 + i;
        Yb[(long long)m * E + e] = acc[j][i] + bias[m];
      }
    }
  }
}

// ---------------------------------------------------------------------------
// Mesh conv with WMMA: G[k=c*5+s, e] built in LDS (dynamic, 83KB max) from
// gathered neighbor columns (L2-resident), then Y = Wp * G.  K = 5*C.
// Block = 256 thr (8 waves), 64 elements/block (4 N-tiles).
// ---------------------------------------------------------------------------
template <int C, int O>
__global__ void meshconv_wmma_kernel(const float* __restrict__ X, long long xbs,
                                     const int* __restrict__ gm,
                                     const __bf16* __restrict__ Wp, const float* __restrict__ bias,
                                     float* __restrict__ Y, long long ybs, int E) {
  constexpr int K = C * 5;
  constexpr int KP = K + 8;
  constexpr int KT = K / 32;
  constexpr int MT = O / 16;
  constexpr int NG = 8 / MT;
  constexpr int NACC = 4 / NG;    // 4 n-tiles per block
  extern __shared__ __bf16 Gs[];

  int b = blockIdx.z;
  int e0 = blockIdx.x * 64;
  const float* Xb = X + (long long)b * xbs;
  for (int idx = threadIdx.x; idx < 64 * C; idx += blockDim.x) {
    int c = idx >> 6, n = idx & 63;
    int e = e0 + n;
    __bf16* g = &Gs[n * KP + c * 5];
    if (e < E) {
      int4 gp = *(const int4*)(gm + ((long long)b * E + e) * 4);
      const float* xc = Xb + (long long)c * E;
      float x0 = xc[e];
      float f1 = xc[gp.x], f2 = xc[gp.y], f3 = xc[gp.z], f4 = xc[gp.w];
      g[0] = (__bf16)x0;
      g[1] = (__bf16)(f1 + f3);
      g[2] = (__bf16)(f2 + f4);
      g[3] = (__bf16)fabsf(f1 - f3);
      g[4] = (__bf16)fabsf(f2 - f4);
    } else {
      g[0] = g[1] = g[2] = g[3] = g[4] = (__bf16)0.f;
    }
  }
  __syncthreads();

  int wave = threadIdx.x >> 5;
  int lane = threadIdx.x & 31;
  int ln = lane & 15, half = lane >> 4;
  int mt = wave % MT, ng = wave / MT;

  v8f acc[NACC];
#pragma unroll
  for (int j = 0; j < NACC; ++j)
#pragma unroll
    for (int i = 0; i < 8; ++i) acc[j][i] = 0.f;

#pragma unroll 4
  for (int kk = 0; kk < KT; ++kk) {
    v16bf A = *(const v16bf*)(Wp + (((long long)mt * KT + kk) * 32 + lane) * 16);
#pragma unroll
    for (int j = 0; j < NACC; ++j) {
      int nt = ng * NACC + j;
      const __bf16* gr = &Gs[(nt * 16 + ln) * KP + kk * 32 + half * 8];
      v16bf Bm = frag16(gr, gr + 16);
      acc[j] = __builtin_amdgcn_wmma_f32_16x16x32_bf16(false, A, false, Bm, (short)0, acc[j], false, false);
    }
  }

  float* Yb = Y + (long long)b * ybs;
#pragma unroll
  for (int j = 0; j < NACC; ++j) {
    int nt = ng * NACC + j;
    int e = e0 + nt * 16 + ln;
    if (e < E) {
#pragma unroll
      for (int i = 0; i < 8; ++i) {
        int m = mt * 16 + half * 8 + i;
        Yb[(long long)m * E + e] = acc[j][i] + bias[m];
      }
    }
  }
}

// softmax([wA,wB]) gate + aggregate into d_out rows 256..383 per batch.
__global__ void agg_kernel(const float* __restrict__ wA, const float* __restrict__ wB,
                           float* __restrict__ out, int E) {
  int row = blockIdx.y;             // b*128 + o
  int b = row >> 7, o = row & 127;
  long long rbase = (long long)row * E;
  long long xe = ((long long)b * 384 + o) * E;
  long long xp = ((long long)b * 384 + 128 + o) * E;
  long long ag = ((long long)b * 384 + 256 + o) * E;
  for (int i = blockIdx.x * blockDim.x + threadIdx.x; i < E; i += gridDim.x * blockDim.x) {
    float a = wA[rbase + i], c = wB[rbase + i];
    float mx = fmaxf(a, c);
    float ea = __expf(a - mx), eb = __expf(c - mx);
    float s0 = ea / (ea + eb);
    out[ag + i] = out[xe + i] * s0 + out[xp + i] * (1.f - s0);
  }
}

// ---------------------------------------------------------------------------
extern "C" void kernel_launch(void* const* d_in, const int* in_sizes, int n_in,
                              void* d_out, int out_size, void* d_ws, size_t ws_size,
                              hipStream_t stream) {
  (void)in_sizes; (void)n_in; (void)out_size; (void)ws_size;
  const float* fe   = (const float*)d_in[0];
  const int*   gmix = (const int*)d_in[1];
  const float* w_e1 = (const float*)d_in[2];
  const float* b_e1 = (const float*)d_in[3];
  const float* w_p1 = (const float*)d_in[4];
  const float* b_p1 = (const float*)d_in[5];
  const float* w_e2 = (const float*)d_in[6];
  const float* b_e2 = (const float*)d_in[7];
  const float* w_p2 = (const float*)d_in[8];
  const float* b_p2 = (const float*)d_in[9];
  const float* wa   = (const float*)d_in[10];
  const float* ba   = (const float*)d_in[11];
  const float* wb   = (const float*)d_in[12];
  const float* bb   = (const float*)d_in[13];
  const float* wal  = (const float*)d_in[14];
  const float* bal  = (const float*)d_in[15];
  const float* wbl  = (const float*)d_in[16];
  const float* bbl  = (const float*)d_in[17];
  const float* wat  = (const float*)d_in[18];
  const float* bat  = (const float*)d_in[19];
  const float* wbt  = (const float*)d_in[20];
  const float* bbt  = (const float*)d_in[21];
  const float* waf  = (const float*)d_in[22];
  const float* baf  = (const float*)d_in[23];
  const float* wbf  = (const float*)d_in[24];
  const float* bbf  = (const float*)d_in[25];
  float* out = (float*)d_out;

  const long long E = NE;
  const long long fN = (long long)NB * 128 * E;
  char* ws = (char*)d_ws;
  size_t off = 0;
  auto carve = [&](size_t bytes) -> void* {
    void* p = ws + off;
    off = (off + bytes + 255) & ~(size_t)255;
    return p;
  };
  float* buf0   = (float*)carve(fN * 4);                      // pre-norm scratch
  float* buf1   = (float*)carve(fN * 4);                      // x1 residual (e then p)
  float* bufAll = (float*)carve((size_t)NB * 64 * E * 4);     // x_all
  float* bufTwo = (float*)carve(fN * 4);                      // x_{a,b}_two
  float* bufWA  = (float*)carve(fN * 4);
  float* bufWB  = (float*)carve(fN * 4);
  float* stats  = (float*)carve((size_t)NB * 128 * 2 * 4);
  __bf16* we2b = (__bf16*)carve(81920 * 2);
  __bf16* wp2b = (__bf16*)carve(81920 * 2);
  __bf16* wab  = (__bf16*)carve(4096 * 2);
  __bf16* wbb  = (__bf16*)carve(4096 * 2);
  __bf16* walb = (__bf16*)carve(4096 * 2);
  __bf16* wblb = (__bf16*)carve(4096 * 2);
  __bf16* watb = (__bf16*)carve(20480 * 2);
  __bf16* wbtb = (__bf16*)carve(20480 * 2);
  __bf16* wafb = (__bf16*)carve(16384 * 2);
  __bf16* wbfb = (__bf16*)carve(16384 * 2);

  auto repack = [&](const float* s, __bf16* d, int O, int K) {
    int n = O * K;
    repack_wmma_a_kernel<<<dim3((n + 255) / 256), dim3(256), 0, stream>>>(s, d, O, K);
  };
  repack(w_e2, we2b, 128, 640); repack(w_p2, wp2b, 128, 640);
  repack(wa,   wab,   32, 128); repack(wb,   wbb,   32, 128);
  repack(wal,  walb,  64,  64); repack(wbl,  wblb,  64,  64);
  repack(wat,  watb,  64, 320); repack(wbt,  wbtb,  64, 320);
  repack(waf,  wafb, 128, 128); repack(wbf,  wbfb, 128, 128);

  const dim3 blk(256);
  const dim3 gE((NE + 255) / 256, 1, NB);
  const dim3 gMesh((NE + 63) / 64, 1, NB);
  const dim3 gConv((NE + 127) / 128, 1, NB);
  const dim3 gNorm((NE + 255) / 256, NB * 128);
  const long long bs128 = 128 * E, bs64 = 64 * E, bs384 = 384 * E;
  const size_t shMesh640 = (size_t)64 * (640 + 8) * 2;   // 82,944 B (<320KB WGP LDS)
  const size_t shMesh320 = (size_t)64 * (320 + 8) * 2;   // 41,984 B

  // ---- edge path ----
  mesh1_kernel<<<gE, blk, 0, stream>>>(fe, gmix, w_e1, b_e1, buf0, 5, 0, NE);
  rowstat_kernel<<<NB * 128, blk, 0, stream>>>(buf0, stats, NE, 1e-5f);
  inorm_act_kernel<<<gNorm, blk, 0, stream>>>(buf0, stats, nullptr, buf1, bs128, NE, 0);
  meshconv_wmma_kernel<128, 128><<<gMesh, blk, shMesh640, stream>>>(buf1, bs128, gmix, we2b, b_e2, buf0, bs128, NE);
  rowstat_kernel<<<NB * 128, blk, 0, stream>>>(buf0, stats, NE, 1e-5f);
  inorm_act_kernel<<<gNorm, blk, 0, stream>>>(buf0, stats, buf1, out, bs384, NE, 1);          // x2_e

  // ---- point path ----
  mesh1_kernel<<<gE, blk, 0, stream>>>(fe, gmix, w_p1, b_p1, buf0, 3, 5, NE);
  rowstat_kernel<<<NB * 128, blk, 0, stream>>>(buf0, stats, NE, 1e-5f);
  inorm_act_kernel<<<gNorm, blk, 0, stream>>>(buf0, stats, nullptr, buf1, bs128, NE, 0);
  meshconv_wmma_kernel<128, 128><<<gMesh, blk, shMesh640, stream>>>(buf1, bs128, gmix, wp2b, b_p2, buf0, bs128, NE);
  rowstat_kernel<<<NB * 128, blk, 0, stream>>>(buf0, stats, NE, 1e-5f);
  inorm_act_kernel<<<gNorm, blk, 0, stream>>>(buf0, stats, buf1, out + 128 * E, bs384, NE, 1); // x2_p

  // ---- x_all = [conv1(x2_e, wa) ; conv1(x2_p, wb)] ----
  conv1_wmma_kernel<128, 32><<<gConv, blk, 0, stream>>>(out, bs384, wab, ba, bufAll, bs64, NE);
  conv1_wmma_kernel<128, 32><<<gConv, blk, 0, stream>>>(out + 128 * E, bs384, wbb, bb, bufAll + 32 * E, bs64, NE);

  // ---- A path: x_a_two -> waf -> inorm -> sigmoid ----
  conv1_wmma_kernel<64, 64><<<gConv, blk, 0, stream>>>(bufAll, bs64, walb, bal, bufTwo, bs128, NE);
  meshconv_wmma_kernel<64, 64><<<gMesh, blk, shMesh320, stream>>>(bufAll, bs64, gmix, watb, bat, bufTwo + 64 * E, bs128, NE);
  conv1_wmma_kernel<128, 128><<<gConv, blk, 0, stream>>>(bufTwo, bs128, wafb, baf, buf0, bs128, NE);
  rowstat_kernel<<<NB * 128, blk, 0, stream>>>(buf0, stats, NE, 1e-5f);
  inorm_act_kernel<<<gNorm, blk, 0, stream>>>(buf0, stats, nullptr, bufWA, bs128, NE, 2);

  // ---- B path ----
  conv1_wmma_kernel<64, 64><<<gConv, blk, 0, stream>>>(bufAll, bs64, wblb, bbl, bufTwo, bs128, NE);
  meshconv_wmma_kernel<64, 64><<<gMesh, blk, shMesh320, stream>>>(bufAll, bs64, gmix, wbtb, bbt, bufTwo + 64 * E, bs128, NE);
  conv1_wmma_kernel<128, 128><<<gConv, blk, 0, stream>>>(bufTwo, bs128, wbfb, bbf, buf0, bs128, NE);
  rowstat_kernel<<<NB * 128, blk, 0, stream>>>(buf0, stats, NE, 1e-5f);
  inorm_act_kernel<<<gNorm, blk, 0, stream>>>(buf0, stats, nullptr, bufWB, bs128, NE, 2);

  // ---- softmax gate + aggregate ----
  agg_kernel<<<gNorm, blk, 0, stream>>>(bufWA, bufWB, out, NE);
}
